// PSE_module_62285615726828
// MI455X (gfx1250) — matrix operand order repaired and checked by reading, not compiled, not run on hardware
//
#include <hip/hip_runtime.h>
#include <cmath>

// ---------------- CDNA5 WMMA types / helpers ----------------
typedef _Float16 half16 __attribute__((ext_vector_type(16)));
typedef float    f32x8  __attribute__((ext_vector_type(8)));

// A-operand (16x32 f16) lane layout: lane l holds row m=l&15; half p maps to
// K = (v>>2)*16 + (l>>4)*8 + (v&3)*2 + (p&1), v = p>>1   (ISA 7.12.2 table)
__device__ __forceinline__ int a_k_of(int p, int hi) {
  int v = p >> 1, h = p & 1;
  return ((v >> 2) << 4) + hi * 8 + ((v & 3) << 1) + h;
}

__device__ __forceinline__ f32x8 wmma16(half16 a, half16 b, f32x8 c) {
  return __builtin_amdgcn_wmma_f32_16x16x32_f16(false, a, false, b, (short)0, c,
                                                false, false);
}

// ---------------- KNN (top-20 nearest incl. self) ----------------
__global__ void __launch_bounds__(256) knn_kernel(const float* __restrict__ x,
                                                  int* __restrict__ idx) {
  __shared__ float px[1024], py[1024], pz[1024], sq[1024];
  int b = blockIdx.x, t = threadIdx.x;
  const float* xb = x + (size_t)b * 3 * 1024;
  for (int i = t; i < 1024; i += 256) {
    float X = xb[i], Y = xb[1024 + i], Z = xb[2048 + i];
    px[i] = X; py[i] = Y; pz[i] = Z; sq[i] = X * X + Y * Y + Z * Z;
  }
  __syncthreads();
  for (int q = t; q < 1024; q += 256) {
    float qx = px[q], qy = py[q], qz = pz[q], qs = sq[q];
    float bd[20]; int bi[20];
    for (int u = 0; u < 20; ++u) { bd[u] = 3.0e38f; bi[u] = 0; }
    for (int j = 0; j < 1024; ++j) {
      float d = qs + sq[j] - 2.0f * (qx * px[j] + qy * py[j] + qz * pz[j]);
      if (d < bd[19]) {
        int pos = 19;
        while (pos > 0 && bd[pos - 1] > d) {
          bd[pos] = bd[pos - 1]; bi[pos] = bi[pos - 1]; --pos;
        }
        bd[pos] = d; bi[pos] = j;
      }
    }
    int* op = idx + ((size_t)b * 1024 + q) * 20;
    for (int u = 0; u < 20; ++u) op[u] = bi[u];
  }
}

// ---------------- gather + LayerNorm + pack to WMMA A layout ----------------
__global__ void __launch_bounds__(256) gather_ln_pack_kernel(
    const float* __restrict__ src, long long bStride, long long rStride,
    long long cStride, const int* __restrict__ idx, const float* __restrict__ g,
    const float* __restrict__ bta, _Float16* __restrict__ Apack, int D, int Dh,
    int KT) {
  __shared__ int idxc[16][20];
  __shared__ float sg[1280], sb[1280];
  __shared__ float part[2][16][16];
  __shared__ float smean[16], srstd[16];
  int t = threadIdx.x;
  int b = blockIdx.x >> 6, tl = blockIdx.x & 63, nb = tl << 4;
  for (int c = t; c < D; c += 256) { sg[c] = g[c]; sb[c] = bta[c]; }
  for (int u = t; u < 320; u += 256)
    idxc[u / 20][u % 20] = idx[((size_t)b * 1024 + nb + u / 20) * 20 + (u % 20)];
  __syncthreads();
  int row = t >> 4, sub = t & 15;
  float s = 0.f, s2 = 0.f;
  for (int col = sub; col < D; col += 16) {
    int kk, dh;
    if (Dh == 64) { kk = col >> 6; dh = col & 63; }
    else          { kk = col / 3;  dh = col - kk * 3; }
    float v = src[(size_t)b * bStride + (long long)idxc[row][kk] * rStride +
                  (long long)dh * cStride];
    s += v; s2 += v * v;
  }
  part[0][row][sub] = s; part[1][row][sub] = s2;
  __syncthreads();
  if (sub == 0) {
    float S = 0.f, S2 = 0.f;
    for (int u = 0; u < 16; ++u) { S += part[0][row][u]; S2 += part[1][row][u]; }
    float mu = S / (float)D;
    smean[row] = mu;
    srstd[row] = rsqrtf(S2 / (float)D - mu * mu + 1e-5f);
  }
  __syncthreads();
  int lane = t & 31, slot = t >> 5;
  int m = lane & 15, hi = lane >> 4;
  for (int kt = slot; kt < KT; kt += 8) {
    half16 v;
    #pragma unroll
    for (int p = 0; p < 16; ++p) {
      int col = kt * 32 + a_k_of(p, hi);
      float val = 0.f;
      if (col < D) {
        int kk, dh;
        if (Dh == 64) { kk = col >> 6; dh = col & 63; }
        else          { kk = col / 3;  dh = col - kk * 3; }
        float xv = src[(size_t)b * bStride + (long long)idxc[m][kk] * rStride +
                       (long long)dh * cStride];
        val = (xv - smean[m]) * srstd[m] * sg[col] + sb[col];
      }
      v[p] = (_Float16)val;
    }
    ((half16*)Apack)[((size_t)blockIdx.x * KT + kt) * 32 + lane] = v;
  }
}

// ---------------- weight repack to WMMA B layout ----------------
__global__ void __launch_bounds__(32) packw_kernel(const float* __restrict__ W,
                                                   _Float16* __restrict__ out,
                                                   int Kreal, int NT, int Ncols) {
  int lane = threadIdx.x;
  int kt = blockIdx.x / NT, nt = blockIdx.x % NT;
  int ncol = nt * 16 + (lane & 15);
  half16 v;
  #pragma unroll
  for (int p = 0; p < 16; ++p) {
    int K = kt * 32 + a_k_of(p, lane >> 4);  // B layout: same K order as A (table order)
    v[p] = (K < Kreal) ? (_Float16)W[(size_t)K * Ncols + ncol] : (_Float16)0.f;
  }
  ((half16*)out)[(size_t)blockIdx.x * 32 + lane] = v;
}

// ---------------- QKV GEMM: [16,Kpad] @ [Kpad,192] per wave ----------------
__global__ void __launch_bounds__(128) qkv_wmma_kernel(
    const _Float16* __restrict__ Apack, const _Float16* __restrict__ Wpack,
    float* __restrict__ qkv, int KT) {
  int lane = threadIdx.x & 31;
  int tile = blockIdx.x * 4 + (threadIdx.x >> 5);
  const half16* Ap = (const half16*)Apack + (size_t)tile * KT * 32;
  const half16* Wp = (const half16*)Wpack;
  f32x8 acc[12];
  #pragma unroll
  for (int i = 0; i < 12; ++i) { f32x8 z = {}; acc[i] = z; }
  for (int kt = 0; kt < KT; ++kt) {
    half16 a = Ap[kt * 32 + lane];
    if (kt + 1 < KT) __builtin_prefetch(&Ap[(kt + 1) * 32 + lane], 0, 1);
    #pragma unroll
    for (int nt = 0; nt < 12; ++nt) {
      half16 bm = Wp[(kt * 12 + nt) * 32 + lane];
      acc[nt] = wmma16(a, bm, acc[nt]);
    }
  }
  int n0 = lane & 15, hi = lane >> 4;
  float* ob = qkv + (size_t)tile * 16 * 192;
  #pragma unroll
  for (int nt = 0; nt < 12; ++nt)
    #pragma unroll
    for (int r = 0; r < 8; ++r)
      ob[(hi * 8 + r) * 192 + nt * 16 + n0] = acc[nt][r];
}

// ---------------- pack Q (scaled), K, V into WMMA operand layouts ----------------
__global__ void __launch_bounds__(32) qkvpack_kernel(
    const float* __restrict__ qkv, _Float16* __restrict__ Qp,
    _Float16* __restrict__ Kp, _Float16* __restrict__ Vp, float scale) {
  int lane = threadIdx.x;
  int t = blockIdx.x & 63, bh = blockIdx.x >> 6;
  int b = bh >> 2, h = bh & 3;
  int m = lane & 15, hi = lane >> 4;
  const float* base = qkv + (size_t)b * 1024 * 192;
  half16 vq, vk;
  int row = t * 16 + m;
  #pragma unroll
  for (int p = 0; p < 16; ++p) {  // Q: A-layout rows=queries, K=d (pad 16->32)
    int d = a_k_of(p, hi);
    vq[p] = (d < 16) ? (_Float16)(base[(size_t)row * 192 + h * 16 + d] * scale)
                     : (_Float16)0.f;
  }
  ((half16*)Qp)[(size_t)blockIdx.x * 32 + lane] = vq;
  #pragma unroll
  for (int p = 0; p < 16; ++p) {  // K^T: B-layout cols=keys, K=d (pad 16->32)
    int d = a_k_of(p, hi);
    vk[p] = (d < 16) ? (_Float16)base[(size_t)row * 192 + 64 + h * 16 + d]
                     : (_Float16)0.f;
  }
  ((half16*)Kp)[(size_t)blockIdx.x * 32 + lane] = vk;
  if (t < 32) {                   // V: B-layout cols=d, K=key-in-chunk(32)
    half16 vv;
    #pragma unroll
    for (int p = 0; p < 16; ++p) {
      int kk = a_k_of(p, hi);
      vv[p] = (_Float16)base[(size_t)(t * 32 + kk) * 192 + 128 + h * 16 + m];
    }
    ((half16*)Vp)[((size_t)bh * 32 + t) * 32 + lane] = vv;
  }
}

// ---------------- attention: scores->LDS, exact softmax, P@V ----------------
__global__ void __launch_bounds__(32) attn_kernel(const _Float16* __restrict__ Qp,
                                                  const _Float16* __restrict__ Kp,
                                                  const _Float16* __restrict__ Vp,
                                                  float* __restrict__ O) {
  __shared__ _Float16 sc[16][1024];
  int lane = threadIdx.x;
  int qt = blockIdx.x & 63, bh = blockIdx.x >> 6;
  int b = bh >> 2, h = bh & 3;
  int n = lane & 15, hi = lane >> 4;
  half16 aq = ((const half16*)Qp)[(size_t)blockIdx.x * 32 + lane];
  const half16* kp = (const half16*)Kp + (size_t)bh * 64 * 32;
  for (int kt = 0; kt < 64; ++kt) {
    f32x8 s = {};
    s = wmma16(aq, kp[kt * 32 + lane], s);
    #pragma unroll
    for (int r = 0; r < 8; ++r) sc[hi * 8 + r][kt * 16 + n] = (_Float16)s[r];
  }
  __syncthreads();
  {
    int row = lane & 15, base = (lane >> 4) * 512;
    float mx = -3.0e38f;
    for (int j = 0; j < 512; ++j) mx = fmaxf(mx, (float)sc[row][base + j]);
    mx = fmaxf(mx, __shfl_xor(mx, 16));
    float sum = 0.f;
    for (int j = 0; j < 512; ++j) sum += expf((float)sc[row][base + j] - mx);
    sum += __shfl_xor(sum, 16);
    float inv = 1.f / sum;
    for (int j = 0; j < 512; ++j)
      sc[row][base + j] = (_Float16)(expf((float)sc[row][base + j] - mx) * inv);
  }
  __syncthreads();
  f32x8 acc = {};
  const half16* vp = (const half16*)Vp + (size_t)bh * 32 * 32;
  for (int c = 0; c < 32; ++c) {
    half16 ap;
    #pragma unroll
    for (int p = 0; p < 16; ++p) ap[p] = sc[lane & 15][c * 32 + a_k_of(p, hi)];
    acc = wmma16(ap, vp[c * 32 + lane], acc);
  }
  float* ob = O + (((size_t)b * 1024 + qt * 16) * 64) + h * 16 + n;
  #pragma unroll
  for (int r = 0; r < 8; ++r) ob[(size_t)(hi * 8 + r) * 64] = acc[r];
}

// ------------- proj + v-skip + LN2 + MLP(GELU) + residual -> concat -------------
__global__ void __launch_bounds__(32) post_kernel(
    const float* __restrict__ Oin, const float* __restrict__ qkv,
    const _Float16* __restrict__ Wproj, const float* __restrict__ bproj,
    const float* __restrict__ g2, const float* __restrict__ b2,
    const _Float16* __restrict__ Wfc1, const float* __restrict__ bfc1,
    const _Float16* __restrict__ Wfc2, const float* __restrict__ bfc2,
    float* __restrict__ outc) {
  __shared__ __align__(32) _Float16 pk[2][32][16];
  int lane = threadIdx.x;
  int m = lane & 15, hi = lane >> 4;
  size_t tile = blockIdx.x;
  const float* Orow = Oin + tile * 16 * 64;
  half16 aO[2];
  #pragma unroll
  for (int kt = 0; kt < 2; ++kt)
    #pragma unroll
    for (int p = 0; p < 16; ++p)
      aO[kt][p] = (_Float16)Orow[m * 64 + kt * 32 + a_k_of(p, hi)];
  const half16* wp = (const half16*)Wproj;
  const float* qrow = qkv + tile * 16 * 192;
  float outv[4][8];
  #pragma unroll
  for (int nt = 0; nt < 4; ++nt) {
    f32x8 acc = {};
    acc = wmma16(aO[0], wp[nt * 32 + lane], acc);
    acc = wmma16(aO[1], wp[(4 + nt) * 32 + lane], acc);
    int c = nt * 16 + m;
    float bb = bproj[c];
    #pragma unroll
    for (int r = 0; r < 8; ++r)
      outv[nt][r] = acc[r] + bb + qrow[(hi * 8 + r) * 192 + 128 + c];
  }
  float mean[8], rstd[8];
  #pragma unroll
  for (int r = 0; r < 8; ++r) {
    float s = 0.f, s2 = 0.f;
    #pragma unroll
    for (int nt = 0; nt < 4; ++nt) { float xx = outv[nt][r]; s += xx; s2 += xx * xx; }
    #pragma unroll
    for (int msk = 1; msk <= 8; msk <<= 1) {
      s += __shfl_xor(s, msk); s2 += __shfl_xor(s2, msk);
    }
    float mu = s * (1.f / 64.f);
    mean[r] = mu;
    rstd[r] = rsqrtf(s2 * (1.f / 64.f) - mu * mu + 1e-5f);
  }
  #pragma unroll
  for (int nt = 0; nt < 4; ++nt) {  // pack LN2(out) into A layout in LDS
    int c = nt * 16 + m;
    float gg = g2[c], bb = b2[c];
    #pragma unroll
    for (int r = 0; r < 8; ++r) {
      int row = hi * 8 + r;
      float h2 = (outv[nt][r] - mean[r]) * rstd[r] * gg + bb;
      int K = c & 31, kt = c >> 5;
      int lp = row + (((K >> 3) & 1) << 4);
      int vv = ((K >> 4) << 2) + ((K & 7) >> 1);
      pk[kt][lp][vv * 2 + (K & 1)] = (_Float16)h2;
    }
  }
  __syncthreads();
  half16 a1[2];
  a1[0] = *(const half16*)pk[0][lane];
  a1[1] = *(const half16*)pk[1][lane];
  __syncthreads();
  const half16* w1 = (const half16*)Wfc1;
  float gel[4][8];
  #pragma unroll
  for (int nt = 0; nt < 4; ++nt) {
    f32x8 acc = {};
    acc = wmma16(a1[0], w1[nt * 32 + lane], acc);
    acc = wmma16(a1[1], w1[(4 + nt) * 32 + lane], acc);
    int c = nt * 16 + m;
    float bb = bfc1[c];
    #pragma unroll
    for (int r = 0; r < 8; ++r) {
      float xx = acc[r] + bb;
      gel[nt][r] = 0.5f * xx * (1.f + erff(xx * 0.70710678118654752f));
    }
  }
  #pragma unroll
  for (int nt = 0; nt < 4; ++nt) {  // pack GELU output
    int c = nt * 16 + m;
    #pragma unroll
    for (int r = 0; r < 8; ++r) {
      int row = hi * 8 + r;
      int K = c & 31, kt = c >> 5;
      int lp = row + (((K >> 3) & 1) << 4);
      int vv = ((K >> 4) << 2) + ((K & 7) >> 1);
      pk[kt][lp][vv * 2 + (K & 1)] = (_Float16)gel[nt][r];
    }
  }
  __syncthreads();
  half16 a2[2];
  a2[0] = *(const half16*)pk[0][lane];
  a2[1] = *(const half16*)pk[1][lane];
  const half16* w2 = (const half16*)Wfc2;
  float* crow = outc + tile * 16 * 512;
  #pragma unroll
  for (int nt = 0; nt < 4; ++nt) {
    f32x8 acc = {};
    acc = wmma16(a2[0], w2[nt * 32 + lane], acc);
    acc = wmma16(a2[1], w2[(4 + nt) * 32 + lane], acc);
    int c = nt * 16 + m;
    float bb = bfc2[c];
    #pragma unroll
    for (int r = 0; r < 8; ++r)
      crow[(size_t)(hi * 8 + r) * 512 + c] = outv[nt][r] + acc[r] + bb;
  }
}

// ---------------- final leaky_relu + LN(512) + transpose ----------------
__global__ void __launch_bounds__(256) final_kernel(
    const float* __restrict__ concat, const float* __restrict__ g,
    const float* __restrict__ bb, float* __restrict__ out) {
  __shared__ float vbuf[16][512];
  __shared__ float ps[2][16][16];
  __shared__ float smean[16], srstd[16];
  int b = blockIdx.x >> 6;
  int nb = (blockIdx.x & 63) << 4;
  int t = threadIdx.x, row = t >> 4, sub = t & 15;
  const float* src = concat + ((size_t)b * 1024 + nb + row) * 512;
  float s = 0.f, s2 = 0.f;
  for (int c = sub; c < 512; c += 16) {
    float xv = src[c];
    xv = xv > 0.f ? xv : 0.01f * xv;
    vbuf[row][c] = xv;
    s += xv; s2 += xv * xv;
  }
  ps[0][row][sub] = s; ps[1][row][sub] = s2;
  __syncthreads();
  if (sub == 0) {
    float S = 0.f, S2 = 0.f;
    for (int u = 0; u < 16; ++u) { S += ps[0][row][u]; S2 += ps[1][row][u]; }
    float mu = S * (1.f / 512.f);
    smean[row] = mu;
    srstd[row] = rsqrtf(S2 * (1.f / 512.f) - mu * mu + 1e-5f);
  }
  __syncthreads();
  for (int c0 = 0; c0 < 512; c0 += 16) {
    int c = c0 + row;
    float val = (vbuf[sub][c] - smean[sub]) * srstd[sub] * g[c] + bb[c];
    out[((size_t)b * 512 + c) * 1024 + nb + sub] = val;
  }
}

// ---------------- host ----------------
extern "C" void kernel_launch(void* const* d_in, const int* in_sizes, int n_in,
                              void* d_out, int out_size, void* d_ws,
                              size_t ws_size, hipStream_t stream) {
  (void)out_size; (void)ws_size;
  // locate x by its unique size (8*3*1024)
  int xi = -1;
  for (int i = 0; i < n_in; ++i)
    if (in_sizes[i] == 8 * 3 * 1024) { xi = i; break; }
  if (xi < 0) xi = 0;
  const float* x = (const float*)d_in[xi];
  const float* P[128]; int sz[128]; int np = 0;
  for (int i = 0; i < n_in && np < 128; ++i)
    if (i != xi) { P[np] = (const float*)d_in[i]; sz[np] = in_sizes[i]; ++np; }
  if (np < 90) return;
  // discriminate sorted-key vs insertion-order flatten of each block dict
  bool sorted = (sz[0] == 64);
  const float *n1g[8], *n1b[8], *qkvw[8], *projw[8], *projb[8];
  const float *n2g[8], *n2b[8], *fc1w[8], *fc1b[8], *fc2w[8], *fc2b[8];
  for (int l = 0; l < 8; ++l) {
    const float* const* q = P + l * 11;
    if (sorted) {
      fc1b[l] = q[0]; fc1w[l] = q[1]; fc2b[l] = q[2]; fc2w[l] = q[3];
      n1b[l] = q[4]; n1g[l] = q[5]; n2b[l] = q[6]; n2g[l] = q[7];
      projb[l] = q[8]; projw[l] = q[9]; qkvw[l] = q[10];
    } else {
      n1g[l] = q[0]; n1b[l] = q[1]; qkvw[l] = q[2]; projw[l] = q[3];
      projb[l] = q[4]; n2g[l] = q[5]; n2b[l] = q[6]; fc1w[l] = q[7];
      fc1b[l] = q[8]; fc2w[l] = q[9]; fc2b[l] = q[10];
    }
  }
  const float* normg = sorted ? P[89] : P[88];
  const float* normb = sorted ? P[88] : P[89];

  // workspace carve-up
  char* w = (char*)d_ws;
  auto alloc = [&](size_t bytes) -> void* {
    void* r = (void*)w;
    w += (bytes + 255) & ~(size_t)255;
    return r;
  };
  int*       idx    = (int*)alloc((size_t)8 * 1024 * 20 * 4);
  _Float16*  Apack  = (_Float16*)alloc((size_t)512 * 40 * 1024);
  float*     qkv    = (float*)alloc((size_t)8 * 1024 * 192 * 4);
  _Float16*  Qp     = (_Float16*)alloc((size_t)8 * 4 * 64 * 1024);
  _Float16*  Kp     = (_Float16*)alloc((size_t)8 * 4 * 64 * 1024);
  _Float16*  Vp     = (_Float16*)alloc((size_t)8 * 4 * 32 * 1024);
  float*     attnO  = (float*)alloc((size_t)8 * 1024 * 64 * 4);
  float*     concat = (float*)alloc((size_t)8 * 1024 * 512 * 4);
  _Float16*  Wq[8];  _Float16* Wpj[8]; _Float16* Wf1[8]; _Float16* Wf2[8];
  for (int l = 0; l < 8; ++l) {
    int KT = l ? 40 : 2;
    Wq[l]  = (_Float16*)alloc((size_t)KT * 12 * 1024);
    Wpj[l] = (_Float16*)alloc((size_t)8 * 1024);
    Wf1[l] = (_Float16*)alloc((size_t)8 * 1024);
    Wf2[l] = (_Float16*)alloc((size_t)8 * 1024);
  }

  // one-time weight repacks (deterministic, replayed each call)
  for (int l = 0; l < 8; ++l) {
    int D = l ? 1280 : 60, KT = l ? 40 : 2;
    packw_kernel<<<dim3(KT * 12), dim3(32), 0, stream>>>(qkvw[l], Wq[l], D, 12, 192);
    packw_kernel<<<dim3(8), dim3(32), 0, stream>>>(projw[l], Wpj[l], 64, 4, 64);
    packw_kernel<<<dim3(8), dim3(32), 0, stream>>>(fc1w[l], Wf1[l], 64, 4, 64);
    packw_kernel<<<dim3(8), dim3(32), 0, stream>>>(fc2w[l], Wf2[l], 64, 4, 64);
  }

  knn_kernel<<<dim3(8), dim3(256), 0, stream>>>(x, idx);

  for (int l = 0; l < 8; ++l) {
    int D = l ? 1280 : 60, Dh = l ? 64 : 3, KT = l ? 40 : 2;
    float scale = 1.0f / sqrtf((float)(D / 4));
    const float* src; long long bS, rS, cS;
    if (l == 0) { src = x;                       bS = 3 * 1024;   rS = 1;   cS = 1024; }
    else        { src = concat + (l - 1) * 64;   bS = 1024 * 512; rS = 512; cS = 1; }
    gather_ln_pack_kernel<<<dim3(512), dim3(256), 0, stream>>>(
        src, bS, rS, cS, idx, n1g[l], n1b[l], Apack, D, Dh, KT);
    qkv_wmma_kernel<<<dim3(128), dim3(128), 0, stream>>>(Apack, Wq[l], qkv, KT);
    qkvpack_kernel<<<dim3(2048), dim3(32), 0, stream>>>(qkv, Qp, Kp, Vp, scale);
    attn_kernel<<<dim3(2048), dim3(32), 0, stream>>>(Qp, Kp, Vp, attnO);
    post_kernel<<<dim3(512), dim3(32), 0, stream>>>(
        attnO, qkv, Wpj[l], projb[l], n2g[l], n2b[l], Wf1[l], fc1b[l], Wf2[l],
        fc2b[l], concat + l * 64);
  }

  final_kernel<<<dim3(512), dim3(256), 0, stream>>>(concat, normg, normb,
                                                    (float*)d_out);
}
